// GATHomGraphClassifier_82033875353650
// MI455X (gfx1250) — matrix operand order repaired and checked by reading, not compiled, not run on hardware
//
#include <hip/hip_runtime.h>

// ---------------- problem constants (from reference) ----------------
#define N_NODES 50000
#define N_EDGES 800000
#define NGRAPH  64
#define HEADS   4
#define CH      64
#define HC      256        // HEADS*CH
#define EDIMC   16
#define NTYPE   8
#define NREL    6
#define NCLS    2
#define NEG_SLOPE 0.2f

typedef __attribute__((ext_vector_type(16))) _Float16 v16h;
typedef __attribute__((ext_vector_type(8)))  float    v8f;

// ---------------- utility zero kernels ----------------
__global__ void k_zero_i32(int* p, int n) {
    int i = blockIdx.x * blockDim.x + threadIdx.x;
    if (i < n) p[i] = 0;
}
__global__ void k_zero_f32(float* p, int n) {
    int i = blockIdx.x * blockDim.x + threadIdx.x;
    if (i < n) p[i] = 0.f;
}

// ---------------- tiny precompute kernels ----------------
// h0_type[t][hc] = sum_k node_emb[t][k] * W0[k][hc]   (8 x 256)
__global__ void k_h0type(const float* __restrict__ node_emb,
                         const float* __restrict__ W0,
                         float* __restrict__ h0t) {
    int idx = blockIdx.x * blockDim.x + threadIdx.x;
    if (idx >= NTYPE * HC) return;
    int t = idx >> 8, hc = idx & 255;
    float s = 0.f;
    for (int k = 0; k < CH; ++k) s += node_emb[t * CH + k] * W0[k * HC + hc];
    h0t[idx] = s;
}

// s_src0_t[t][h], s_dst0_t[t][h]  (8 x 4 each)
__global__ void k_s0type(const float* __restrict__ h0t,
                         const float* __restrict__ as, const float* __restrict__ ad,
                         float* __restrict__ ssrc, float* __restrict__ sdst) {
    int idx = threadIdx.x;
    if (idx >= NTYPE * HEADS) return;
    int t = idx >> 2, h = idx & 3;
    float ss = 0.f, sd = 0.f;
    for (int c = 0; c < CH; ++c) {
        float v = h0t[t * HC + h * CH + c];
        ss += v * as[h * CH + c];
        sd += v * ad[h * CH + c];
    }
    ssrc[idx] = ss; sdst[idx] = sd;
}

// s_edge_t[l][r][h] = sum_c (edge_emb[r] @ We_l)[h*64+c] * att_edge_l[h][c]
__global__ void k_sedge(const float* __restrict__ ee,
                        const float* __restrict__ We0, const float* __restrict__ We1,
                        const float* __restrict__ ae0, const float* __restrict__ ae1,
                        float* __restrict__ se0, float* __restrict__ se1) {
    int idx = threadIdx.x;
    if (idx >= 2 * NREL * HEADS) return;
    int l = idx / (NREL * HEADS);
    int rem = idx % (NREL * HEADS);
    int r = rem >> 2, h = rem & 3;
    const float* We = l ? We1 : We0;
    const float* ae = l ? ae1 : ae0;
    float sum = 0.f;
    for (int c = 0; c < CH; ++c) {
        float t = 0.f;
        for (int k = 0; k < EDIMC; ++k) t += ee[r * EDIMC + k] * We[k * HC + h * CH + c];
        sum += t * ae[h * CH + c];
    }
    (l ? se1 : se0)[r * HEADS + h] = sum;
}

// convert W1 (f32 [64,256]) -> f16
__global__ void k_wconv(const float* __restrict__ W, _Float16* __restrict__ Wh, int n) {
    int i = blockIdx.x * blockDim.x + threadIdx.x;
    if (i < n) Wh[i] = (_Float16)W[i];
}

// ---------------- CSR-by-dst construction ----------------
__global__ void k_hist(const int* __restrict__ dstA, int* __restrict__ counts) {
    int e = blockIdx.x * blockDim.x + threadIdx.x;
    if (e < N_EDGES) atomicAdd(&counts[dstA[e]], 1);
}

// single-block exclusive scan over counts[N] -> row_ptr[N+1]; also seeds cursor
__global__ void k_scan(const int* __restrict__ counts,
                       int* __restrict__ row_ptr, int* __restrict__ cursor) {
    __shared__ int sh[256];
    __shared__ int carry;
    if (threadIdx.x == 0) carry = 0;
    __syncthreads();
    for (int base = 0; base < N_NODES; base += 256) {
        int i = base + threadIdx.x;
        int v = (i < N_NODES) ? counts[i] : 0;
        sh[threadIdx.x] = v;
        __syncthreads();
        for (int off = 1; off < 256; off <<= 1) {
            int t = (threadIdx.x >= off) ? sh[threadIdx.x - off] : 0;
            __syncthreads();
            sh[threadIdx.x] += t;
            __syncthreads();
        }
        int excl = sh[threadIdx.x] - v;
        if (i < N_NODES) {
            int rp = carry + excl;
            row_ptr[i] = rp;
            cursor[i]  = rp;
        }
        __syncthreads();
        if (threadIdx.x == 255) carry += sh[255];
        __syncthreads();
    }
    if (threadIdx.x == 0) row_ptr[N_NODES] = carry;   // == E
}

__global__ void k_fill(const int* __restrict__ dstA, int* __restrict__ cursor,
                       int* __restrict__ edge_by_dst) {
    int e = blockIdx.x * blockDim.x + threadIdx.x;
    if (e < N_EDGES) {
        int p = atomicAdd(&cursor[dstA[e]], 1);
        edge_by_dst[p] = e;
    }
}

// ---------------- per-edge attention logits (stored at CSR-sorted position) ----------------
__global__ void k_edge_logits(const int* __restrict__ edge_by_dst,
                              const int* __restrict__ srcA, const int* __restrict__ dstA,
                              const int* __restrict__ etype, const int* __restrict__ ntype,
                              const float* __restrict__ ssrc, const float* __restrict__ sdst,
                              const float* __restrict__ sedge_t,
                              int use_type, float* __restrict__ alphaL) {
    int p = blockIdx.x * blockDim.x + threadIdx.x;
    if (p >= N_EDGES) return;
    int e = edge_by_dst[p];
    int s = srcA[e], d = dstA[e], t = etype[e];
    const float* ps = use_type ? (ssrc + ntype[s] * HEADS) : (ssrc + (size_t)s * HEADS);
    const float* pd = use_type ? (sdst + ntype[d] * HEADS) : (sdst + (size_t)d * HEADS);
    const float* pe = sedge_t + t * HEADS;
#pragma unroll
    for (int h = 0; h < HEADS; ++h) {
        float a = ps[h] + pd[h] + pe[h];
        alphaL[(size_t)p * HEADS + h] = (a > 0.f) ? a : NEG_SLOPE * a;
    }
}

// ---------------- wave-per-node softmax + weighted aggregation ----------------
// Lane L owns (h=j/2, c=L+32*(j&1)) for j=0..7 -> idx = L + 32*j in the 256-wide row.
__global__ void k_aggregate(const int* __restrict__ row_ptr,
                            const int* __restrict__ edge_by_dst,
                            const float* __restrict__ alphaL,   // [E][4] in CSR order
                            const int* __restrict__ srcA,
                            const int* __restrict__ node_type,
                            const float* __restrict__ hsrcBase, // h0_type or h1
                            const float* __restrict__ bias,     // [64]
                            int layer0,
                            float* __restrict__ x_out,          // layer0: x1 [N,64]
                            const int* __restrict__ batch,      // layer1 pooling
                            float* __restrict__ gsum, float* __restrict__ gcnt) {
    int wave = blockIdx.x * (blockDim.x >> 5) + (threadIdx.x >> 5);
    int lane = threadIdx.x & 31;
    if (wave >= N_NODES) return;
    int start = row_ptr[wave], end = row_ptr[wave + 1];

    // pass 1: segment max per head (lanes split edges; max is order-free)
    float m0 = -3.0e38f, m1 = -3.0e38f, m2 = -3.0e38f, m3 = -3.0e38f;
    for (int i = start + lane; i < end; i += 32) {
        const float* lg = alphaL + (size_t)i * HEADS;
        m0 = fmaxf(m0, lg[0]); m1 = fmaxf(m1, lg[1]);
        m2 = fmaxf(m2, lg[2]); m3 = fmaxf(m3, lg[3]);
    }
#pragma unroll
    for (int off = 16; off > 0; off >>= 1) {
        m0 = fmaxf(m0, __shfl_xor(m0, off));
        m1 = fmaxf(m1, __shfl_xor(m1, off));
        m2 = fmaxf(m2, __shfl_xor(m2, off));
        m3 = fmaxf(m3, __shfl_xor(m3, off));
    }

    // pass 2: all lanes walk all edges; each lane accumulates its 8 features
    float d0 = 0.f, d1 = 0.f, d2 = 0.f, d3 = 0.f;
    float acc[8];
#pragma unroll
    for (int j = 0; j < 8; ++j) acc[j] = 0.f;
    for (int i = start; i < end; ++i) {
        const float* lg = alphaL + (size_t)i * HEADS;
        float w0 = __expf(lg[0] - m0), w1 = __expf(lg[1] - m1);
        float w2 = __expf(lg[2] - m2), w3 = __expf(lg[3] - m3);
        d0 += w0; d1 += w1; d2 += w2; d3 += w3;
        int e = edge_by_dst[i];
        int s = srcA[e];
        const float* hs = layer0 ? (hsrcBase + node_type[s] * HC)
                                 : (hsrcBase + (size_t)s * HC);
        acc[0] += w0 * hs[lane];        acc[1] += w0 * hs[lane + 32];
        acc[2] += w1 * hs[lane + 64];   acc[3] += w1 * hs[lane + 96];
        acc[4] += w2 * hs[lane + 128];  acc[5] += w2 * hs[lane + 160];
        acc[6] += w3 * hs[lane + 192];  acc[7] += w3 * hs[lane + 224];
    }
    float i0 = 1.f / (d0 + 1e-16f), i1 = 1.f / (d1 + 1e-16f);
    float i2 = 1.f / (d2 + 1e-16f), i3 = 1.f / (d3 + 1e-16f);
    // head mean: lane L holds all 4 heads for c=L (even j) and c=L+32 (odd j)
    float mean0 = 0.25f * (acc[0] * i0 + acc[2] * i1 + acc[4] * i2 + acc[6] * i3);
    float mean1 = 0.25f * (acc[1] * i0 + acc[3] * i1 + acc[5] * i2 + acc[7] * i3);
    float v0 = mean0 + bias[lane];
    float v1 = mean1 + bias[lane + 32];
    v0 = (v0 > 0.f) ? v0 : (__expf(v0) - 1.f);   // ELU
    v1 = (v1 > 0.f) ? v1 : (__expf(v1) - 1.f);
    if (layer0) {
        x_out[(size_t)wave * CH + lane]      = v0;
        x_out[(size_t)wave * CH + lane + 32] = v1;
    } else {
        int b = batch[wave];
        atomicAdd(gsum + b * CH + lane,      v0);
        atomicAdd(gsum + b * CH + lane + 32, v1);
        if (lane == 0) atomicAdd(gcnt + b, 1.0f);
    }
}

// ---------------- WMMA GEMM: h1 = x1 @ W1   ([50000,64] x [64,256]) ----------------
// One wave per 16x16 output tile, K=64 via two v_wmma_f32_16x16x32_f16 steps.
// A layout (ISA 7.12.2, 16-bit A 16x32): lane g=L/16, m=L%16;
//   halves t=0..7  -> K = 8g + t,  halves t=8..15 -> K = 16 + 8g + (t-8).
// B layout (32x16 f16, same K order as A): halves t=0..7 -> K = 8g + t, t=8..15 -> K = 16 + 8g + (t-8), n = L%16.
// D layout: VGPR v -> row = v + 8g, col = L%16.
__global__ void k_gemm_h1(const float* __restrict__ x1,
                          const _Float16* __restrict__ Wh,
                          float* __restrict__ h1) {
    int wave = blockIdx.x * (blockDim.x >> 5) + (threadIdx.x >> 5);
    const int MT = N_NODES / 16;           // 3125 exact
    if (wave >= MT * 16) return;
    int tm = wave >> 4;
    int tn = wave & 15;
    int lane = threadIdx.x & 31;
    int g = lane >> 4;
    int r = lane & 15;
    const float* arow = x1 + (size_t)(tm * 16 + r) * CH;   // row m = r for both halves
    v8f c = {};
#pragma unroll
    for (int kk = 0; kk < 2; ++kk) {
        int koff = kk * 32;
        v16h a, b;
#pragma unroll
        for (int t = 0; t < 8; ++t) {
            a[t]     = (_Float16)arow[koff + 8 * g + t];
            a[t + 8] = (_Float16)arow[koff + 16 + 8 * g + t];
        }
#pragma unroll
        for (int t = 0; t < 16; ++t) {
            b[t] = Wh[(size_t)(koff + 8 * g + (t & 7) + ((t >> 3) << 4)) * HC + tn * 16 + r];
        }
        c = __builtin_amdgcn_wmma_f32_16x16x32_f16(false, a, false, b,
                                                   (short)0, c, false, false);
    }
#pragma unroll
    for (int v = 0; v < 8; ++v) {
        h1[(size_t)(tm * 16 + v + 8 * g) * HC + tn * 16 + r] = c[v];
    }
}

// ---------------- per-node attention scalars for layer 1 ----------------
__global__ void k_s1(const float* __restrict__ h1,
                     const float* __restrict__ att_src, const float* __restrict__ att_dst,
                     float* __restrict__ s_src, float* __restrict__ s_dst) {
    int wave = blockIdx.x * (blockDim.x >> 5) + (threadIdx.x >> 5);
    int lane = threadIdx.x & 31;
    if (wave >= N_NODES) return;
    const float* hv = h1 + (size_t)wave * HC;
    float ps[4] = {0.f, 0.f, 0.f, 0.f};
    float pd[4] = {0.f, 0.f, 0.f, 0.f};
#pragma unroll
    for (int j = 0; j < 8; ++j) {
        int idx = lane + 32 * j;          // h = j/2
        float v = hv[idx];
        ps[j >> 1] += v * att_src[idx];
        pd[j >> 1] += v * att_dst[idx];
    }
#pragma unroll
    for (int off = 16; off > 0; off >>= 1) {
#pragma unroll
        for (int h = 0; h < 4; ++h) {
            ps[h] += __shfl_xor(ps[h], off);
            pd[h] += __shfl_xor(pd[h], off);
        }
    }
    if (lane == 0) {
#pragma unroll
        for (int h = 0; h < 4; ++h) {
            s_src[(size_t)wave * HEADS + h] = ps[h];
            s_dst[(size_t)wave * HEADS + h] = pd[h];
        }
    }
}

// ---------------- classifier: relu(g@cw1+cb1)@cw2+cb2 ----------------
__global__ void k_cls(const float* __restrict__ gsum, const float* __restrict__ gcnt,
                      const float* __restrict__ cw1, const float* __restrict__ cb1,
                      const float* __restrict__ cw2, const float* __restrict__ cb2,
                      float* __restrict__ out) {
    int gI = threadIdx.x;
    if (gI >= NGRAPH) return;
    float inv = 1.f / fmaxf(gcnt[gI], 1.0f);
    float o0 = cb2[0], o1 = cb2[1];
    for (int k = 0; k < CH; ++k) {
        float hsum = cb1[k];
        for (int c = 0; c < CH; ++c) hsum += gsum[gI * CH + c] * inv * cw1[c * CH + k];
        hsum = fmaxf(hsum, 0.f);
        o0 += hsum * cw2[k * NCLS + 0];
        o1 += hsum * cw2[k * NCLS + 1];
    }
    out[gI * NCLS + 0] = o0;
    out[gI * NCLS + 1] = o1;
}

// ---------------- launcher ----------------
extern "C" void kernel_launch(void* const* d_in, const int* in_sizes, int n_in,
                              void* d_out, int out_size, void* d_ws, size_t ws_size,
                              hipStream_t stream) {
    const int*   node_type  = (const int*)  d_in[0];
    const int*   edge_type  = (const int*)  d_in[1];
    const int*   edge_index = (const int*)  d_in[2];
    const int*   batch      = (const int*)  d_in[3];
    const float* node_emb   = (const float*)d_in[4];
    const float* edge_emb   = (const float*)d_in[5];
    const float* W0         = (const float*)d_in[6];
    const float* We0        = (const float*)d_in[7];
    const float* att_src0   = (const float*)d_in[8];
    const float* att_dst0   = (const float*)d_in[9];
    const float* att_edge0  = (const float*)d_in[10];
    const float* b0         = (const float*)d_in[11];
    const float* W1         = (const float*)d_in[12];
    const float* We1        = (const float*)d_in[13];
    const float* att_src1   = (const float*)d_in[14];
    const float* att_dst1   = (const float*)d_in[15];
    const float* att_edge1  = (const float*)d_in[16];
    const float* b1         = (const float*)d_in[17];
    const float* cw1        = (const float*)d_in[18];
    const float* cb1        = (const float*)d_in[19];
    const float* cw2        = (const float*)d_in[20];
    const float* cb2        = (const float*)d_in[21];
    const int* srcA = edge_index;
    const int* dstA = edge_index + N_EDGES;
    float* out = (float*)d_out;

    // workspace layout (256B aligned)
    char* ws = (char*)d_ws;
    size_t off = 0;
    auto take = [&](size_t bytes) -> char* {
        char* p = ws + off;
        off = (off + bytes + 255) & ~(size_t)255;
        return p;
    };
    int*   counts      = (int*)  take((size_t)N_NODES * 4);
    int*   row_ptr     = (int*)  take((size_t)(N_NODES + 1) * 4);
    int*   cursor      = (int*)  take((size_t)N_NODES * 4);
    int*   edge_by_dst = (int*)  take((size_t)N_EDGES * 4);
    float* alphaL      = (float*)take((size_t)N_EDGES * HEADS * 4);
    float* h0t         = (float*)take((size_t)NTYPE * HC * 4);
    float* s_src0_t    = (float*)take((size_t)NTYPE * HEADS * 4);
    float* s_dst0_t    = (float*)take((size_t)NTYPE * HEADS * 4);
    float* s_e0        = (float*)take((size_t)NREL * HEADS * 4);
    float* s_e1        = (float*)take((size_t)NREL * HEADS * 4);
    _Float16* Wh       = (_Float16*)take((size_t)CH * HC * 2);
    float* x1          = (float*)take((size_t)N_NODES * CH * 4);
    float* h1          = (float*)take((size_t)N_NODES * HC * 4);
    float* s_src1      = (float*)take((size_t)N_NODES * HEADS * 4);
    float* s_dst1      = (float*)take((size_t)N_NODES * HEADS * 4);
    float* gsum        = (float*)take((size_t)NGRAPH * CH * 4);
    float* gcnt        = (float*)take((size_t)NGRAPH * 4);
    (void)ws_size; (void)in_sizes; (void)n_in; (void)out_size;

    const int TB = 256;
    int gridN  = (N_NODES + TB - 1) / TB;          // 196
    int gridE  = (N_EDGES + TB - 1) / TB;          // 3125
    int gridWN = (N_NODES + 7) / 8;                // 6250 (8 waves/block, wave/node)

    // zero accumulation buffers (harness does not re-poison between replays)
    k_zero_i32<<<gridN, TB, 0, stream>>>(counts, N_NODES);
    k_zero_f32<<<(NGRAPH * CH + TB - 1) / TB, TB, 0, stream>>>(gsum, NGRAPH * CH);
    k_zero_f32<<<1, NGRAPH, 0, stream>>>(gcnt, NGRAPH);

    // tiny precomputes
    k_h0type<<<(NTYPE * HC + TB - 1) / TB, TB, 0, stream>>>(node_emb, W0, h0t);
    k_s0type<<<1, 64, 0, stream>>>(h0t, att_src0, att_dst0, s_src0_t, s_dst0_t);
    k_sedge<<<1, 64, 0, stream>>>(edge_emb, We0, We1, att_edge0, att_edge1, s_e0, s_e1);
    k_wconv<<<(CH * HC + TB - 1) / TB, TB, 0, stream>>>(W1, Wh, CH * HC);

    // CSR by destination
    k_hist<<<gridE, TB, 0, stream>>>(dstA, counts);
    k_scan<<<1, 256, 0, stream>>>(counts, row_ptr, cursor);
    k_fill<<<gridE, TB, 0, stream>>>(dstA, cursor, edge_by_dst);

    // ---- layer 0 (h via 8-row type table) ----
    k_edge_logits<<<gridE, TB, 0, stream>>>(edge_by_dst, srcA, dstA, edge_type, node_type,
                                            s_src0_t, s_dst0_t, s_e0, 1, alphaL);
    k_aggregate<<<gridWN, TB, 0, stream>>>(row_ptr, edge_by_dst, alphaL, srcA, node_type,
                                           h0t, b0, 1, x1, batch, gsum, gcnt);

    // ---- layer 1 ----
    k_gemm_h1<<<gridWN, TB, 0, stream>>>(x1, Wh, h1);   // 50000 tiles, wave/tile (WMMA)
    k_s1<<<gridWN, TB, 0, stream>>>(h1, att_src1, att_dst1, s_src1, s_dst1);
    k_edge_logits<<<gridE, TB, 0, stream>>>(edge_by_dst, srcA, dstA, edge_type, node_type,
                                            s_src1, s_dst1, s_e1, 0, alphaL);
    k_aggregate<<<gridWN, TB, 0, stream>>>(row_ptr, edge_by_dst, alphaL, srcA, node_type,
                                           h1, b1, 0, nullptr, batch, gsum, gcnt);

    // ---- pooled classifier ----
    k_cls<<<1, NGRAPH, 0, stream>>>(gsum, gcnt, cw1, cb1, cw2, cb2, out);
}
